// SRDELayer_87978110091730
// MI455X (gfx1250) — compile-verified, matrix operand order checked by reading, not run on hardware
//
#include <hip/hip_runtime.h>
#include <hip/hip_bf16.h>

typedef __bf16 bf16_t;
typedef __attribute__((ext_vector_type(16))) __bf16 v16bf;
typedef __attribute__((ext_vector_type(8)))  __bf16 v8bf;
typedef __attribute__((ext_vector_type(8)))  float  v8f;

#define T_TOK   4096
#define H_DIM   2048
#define FF_DIM  8192
#define E_NUM   8
#define A_NUM   64
#define NS_NUM  5033
#define N1      16777216        // FF*H
#define P_NUM   50331648        // 3*FF*H
#define SEL_PAD 8192
#define TIE_CAP 16384
#define LDSS    40              // padded LDS row stride (elems), 80B = 16B-aligned chunks

// d_out layout (floats): out | router_logits | router_weights | top_idx | aux
#define LOG_OFF 8388608
#define RWT_OFF 8421376
#define TID_OFF 8429568
#define AUX_OFF 8437760

// ---------- helpers ----------
__device__ __forceinline__ bf16_t f2bf(float f) {
  unsigned u = __float_as_uint(f);
  unsigned r = 0x7FFFu + ((u >> 16) & 1u);
  unsigned short h = (unsigned short)((u + r) >> 16);
  return __builtin_bit_cast(bf16_t, h);
}
__device__ __forceinline__ unsigned fkey(float v) {   // order-preserving map
  unsigned u = __float_as_uint(v);
  return (u & 0x80000000u) ? ~u : (u | 0x80000000u);
}
union FragBF { v16bf v; v8bf h[2]; };

// ---------- router: logits, top-2 softmax, stats ----------
__global__ __launch_bounds__(256) void router_kernel(
    const float* __restrict__ X, const float* __restrict__ Rw,
    float* __restrict__ dout, float* __restrict__ stat) {
  int wave = threadIdx.x >> 5, lane = threadIdx.x & 31;
  int t = blockIdx.x * 8 + wave;
  const float* xr = X + (size_t)t * H_DIM;
  float acc[E_NUM] = {};
  for (int h = lane; h < H_DIM; h += 32) {
    float xv = xr[h];
#pragma unroll
    for (int e = 0; e < E_NUM; ++e) acc[e] += xv * Rw[e * H_DIM + h];
  }
#pragma unroll
  for (int e = 0; e < E_NUM; ++e)
#pragma unroll
    for (int off = 16; off; off >>= 1) acc[e] += __shfl_xor(acc[e], off, 32);
  if (lane == 0) {
    float lg[E_NUM];
#pragma unroll
    for (int e = 0; e < E_NUM; ++e) {
      lg[e] = fminf(fmaxf(acc[e], -50.f), 50.f);
      dout[LOG_OFF + (size_t)t * E_NUM + e] = lg[e];
    }
    int i0 = 0;
    for (int e = 1; e < E_NUM; ++e) if (lg[e] > lg[i0]) i0 = e;
    int i1 = (i0 == 0) ? 1 : 0;
    for (int e = 0; e < E_NUM; ++e) if (e != i0 && lg[e] > lg[i1]) i1 = e;
    float m = lg[i0];
    float e0 = __expf(lg[i0] - m), e1 = __expf(lg[i1] - m);
    float w0 = e0 / (e0 + e1), w1 = e1 / (e0 + e1);
    dout[RWT_OFF + t * 2 + 0] = w0;  dout[RWT_OFF + t * 2 + 1] = w1;
    dout[TID_OFF + t * 2 + 0] = (float)i0;  dout[TID_OFF + t * 2 + 1] = (float)i1;
    float mm = lg[0];
    for (int e = 1; e < E_NUM; ++e) mm = fmaxf(mm, lg[e]);
    float pe[E_NUM], s = 0.f;
#pragma unroll
    for (int e = 0; e < E_NUM; ++e) { pe[e] = __expf(lg[e] - mm); s += pe[e]; }
#pragma unroll
    for (int e = 0; e < E_NUM; ++e) atomicAdd(&stat[e], pe[e] / s);   // sum probs
    atomicAdd(&stat[8 + i0], 1.f);  atomicAdd(&stat[8 + i1], 1.f);    // counts
    atomicAdd(&stat[16 + i0], w0);  atomicAdd(&stat[16 + i1], w1);    // coeff
  }
}

__global__ void aux_kernel(const float* __restrict__ stat, float* __restrict__ dout) {
  float a = 0.f;
  for (int e = 0; e < E_NUM; ++e)
    a += (stat[e] / (float)T_TOK) * (stat[8 + e] / (float)(T_TOK * 2));
  dout[AUX_OFF] = (float)E_NUM * a;
}

// ---------- top-NS selection over mask_logits ----------
__global__ __launch_bounds__(256) void hist1_kernel(const float* __restrict__ ml,
                                                    unsigned* __restrict__ hist) {
  __shared__ unsigned lh[2048];
  for (int i = threadIdx.x; i < 2048; i += 256) lh[i] = 0u;
  __syncthreads();
  for (size_t i = (size_t)blockIdx.x * 256 + threadIdx.x; i < (size_t)P_NUM;
       i += (size_t)gridDim.x * 256)
    atomicAdd(&lh[fkey(ml[i]) >> 21], 1u);
  __syncthreads();
  for (int i = threadIdx.x; i < 2048; i += 256)
    if (lh[i]) atomicAdd(&hist[i], lh[i]);
}

__global__ void thresh1_kernel(const unsigned* __restrict__ hist, int* __restrict__ ctl) {
  unsigned cum = 0;
  for (int b = 2047; b >= 0; --b) {
    cum += hist[b];
    if (cum >= (unsigned)NS_NUM) {
      ctl[2] = b;                             // B1
      ctl[3] = NS_NUM - (int)(cum - hist[b]); // rem1
      return;
    }
  }
}

__global__ __launch_bounds__(256) void hist2_kernel(const float* __restrict__ ml,
                                                    const int* __restrict__ ctl,
                                                    unsigned* __restrict__ hist) {
  __shared__ unsigned lh[2048];
  for (int i = threadIdx.x; i < 2048; i += 256) lh[i] = 0u;
  __syncthreads();
  unsigned B1 = (unsigned)ctl[2];
  for (size_t i = (size_t)blockIdx.x * 256 + threadIdx.x; i < (size_t)P_NUM;
       i += (size_t)gridDim.x * 256) {
    unsigned k = fkey(ml[i]);
    if ((k >> 21) == B1) atomicAdd(&lh[(k >> 10) & 0x7FFu], 1u);
  }
  __syncthreads();
  for (int i = threadIdx.x; i < 2048; i += 256)
    if (lh[i]) atomicAdd(&hist[i], lh[i]);
}

__global__ void thresh2_kernel(const unsigned* __restrict__ hist, int* __restrict__ ctl) {
  unsigned cum = 0, tgt = (unsigned)ctl[3];
  for (int b = 2047; b >= 0; --b) {
    cum += hist[b];
    if (cum >= tgt) {
      ctl[4] = b;                              // B2
      ctl[5] = (int)(tgt - (cum - hist[b]));   // rem2
      return;
    }
  }
}

__global__ __launch_bounds__(256) void select_kernel(
    const float* __restrict__ ml, int* __restrict__ ctl,
    unsigned long long* __restrict__ sel, unsigned long long* __restrict__ tie) {
  unsigned B1 = (unsigned)ctl[2], B2 = (unsigned)ctl[4];
  for (size_t i = (size_t)blockIdx.x * 256 + threadIdx.x; i < (size_t)P_NUM;
       i += (size_t)gridDim.x * 256) {
    unsigned k = fkey(ml[i]);
    unsigned b1 = k >> 21;
    if (b1 < B1) continue;
    unsigned long long comb = ((unsigned long long)k << 32) | (unsigned)(~(unsigned)i);
    if (b1 > B1) {
      sel[atomicAdd(&ctl[0], 1)] = comb;
    } else {
      unsigned b2 = (k >> 10) & 0x7FFu;
      if (b2 > B2) sel[atomicAdd(&ctl[0], 1)] = comb;
      else if (b2 == B2) {
        int p = atomicAdd(&ctl[1], 1);
        if (p < TIE_CAP) tie[p] = comb;
      }
    }
  }
}

__device__ void bitonic_desc(unsigned long long* d, int n) {
  for (int k = 2; k <= n; k <<= 1)
    for (int j = k >> 1; j > 0; j >>= 1) {
      for (int i = threadIdx.x; i < n; i += blockDim.x) {
        int l = i ^ j;
        if (l > i) {
          unsigned long long a = d[i], b = d[l];
          bool seg = ((i & k) == 0);           // descending segment
          if ((seg && a < b) || (!seg && a > b)) { d[i] = b; d[l] = a; }
        }
      }
      __syncthreads();
    }
}

__global__ __launch_bounds__(1024) void tie_sort_kernel(
    unsigned long long* __restrict__ tie, unsigned long long* __restrict__ sel,
    const int* __restrict__ ctl) {
  int tc = ctl[1]; if (tc > TIE_CAP) tc = TIE_CAP;
  int rem2 = ctl[5], sc = ctl[0];
  for (int i = threadIdx.x; i < TIE_CAP; i += 1024) if (i >= tc) tie[i] = 0ull;
  __syncthreads();
  bitonic_desc(tie, TIE_CAP);
  for (int j = threadIdx.x; j < rem2; j += 1024) sel[sc + j] = tie[j];
}

__global__ __launch_bounds__(1024) void sel_sort_kernel(unsigned long long* __restrict__ sel) {
  for (int i = threadIdx.x; i < SEL_PAD; i += 1024) if (i >= NS_NUM) sel[i] = 0ull;
  __syncthreads();
  bitonic_desc(sel, SEL_PAD);
}

// ---------- deltas ----------
__global__ void atom_softmax_kernel(const float* __restrict__ eaw, float* __restrict__ aw) {
  int e = threadIdx.x;
  if (e >= E_NUM) return;
  float m = eaw[e * A_NUM];
  for (int a = 1; a < A_NUM; ++a) m = fmaxf(m, eaw[e * A_NUM + a]);
  float s = 0.f;
  for (int a = 0; a < A_NUM; ++a) s += __expf(eaw[e * A_NUM + a] - m);
  for (int a = 0; a < A_NUM; ++a) aw[e * A_NUM + a] = __expf(eaw[e * A_NUM + a] - m) / s;
}

__global__ __launch_bounds__(256) void wd_kernel(
    const float* __restrict__ atoms, const float* __restrict__ aw,
    const float* __restrict__ imp, float* __restrict__ wd) {
  int i = blockIdx.x * 256 + threadIdx.x;
  if (i >= E_NUM * NS_NUM) return;
  int e = i / NS_NUM, n = i - e * NS_NUM;
  float s = 0.f;
  for (int a = 0; a < A_NUM; ++a) s += aw[e * A_NUM + a] * atoms[(size_t)a * NS_NUM + n];
  float sg = 1.f / (1.f + __expf(-imp[i]));
  wd[i] = s * sg;
}

__global__ __launch_bounds__(256) void md_kernel(const float* __restrict__ wd,
                                                 const float* __restrict__ stat,
                                                 float* __restrict__ md) {
  int i = blockIdx.x * 256 + threadIdx.x;
  if (i >= NS_NUM) return;
  float s = 0.f;
#pragma unroll
  for (int e = 0; e < E_NUM; ++e) s += stat[16 + e] * wd[e * NS_NUM + i];
  md[i] = s / (float)T_TOK;
}

// ---------- bf16 conversion + sparse weight fixup ----------
__global__ __launch_bounds__(256) void convert_x_kernel(const float* __restrict__ x,
                                                        bf16_t* __restrict__ xb) {
  for (size_t i = (size_t)blockIdx.x * 256 + threadIdx.x; i < (size_t)T_TOK * H_DIM;
       i += (size_t)gridDim.x * 256)
    xb[i] = f2bf(x[i]);
}

__global__ __launch_bounds__(256) void convert_w_kernel(
    const float* __restrict__ g, const float* __restrict__ u,
    const float* __restrict__ d, bf16_t* __restrict__ wb) {
  for (size_t i = (size_t)blockIdx.x * 256 + threadIdx.x; i < (size_t)P_NUM;
       i += (size_t)gridDim.x * 256) {
    float v = (i < (size_t)N1) ? g[i]
            : (i < (size_t)2 * N1) ? u[i - N1] : d[i - (size_t)2 * N1];
    wb[i] = f2bf(v);
  }
}

__global__ __launch_bounds__(256) void fixup_kernel(
    const unsigned long long* __restrict__ sel, const float* __restrict__ md,
    const float* __restrict__ g, const float* __restrict__ u,
    const float* __restrict__ d, bf16_t* __restrict__ wb) {
  int i = blockIdx.x * 256 + threadIdx.x;
  if (i >= NS_NUM) return;
  size_t f = (size_t)(~(unsigned)(sel[i] & 0xFFFFFFFFull));
  float orig = (f < (size_t)N1) ? g[f]
             : (f < (size_t)2 * N1) ? u[f - N1] : d[f - (size_t)2 * N1];
  wb[f] = f2bf(orig + md[i]);
}

// ---------- WMMA GEMM 1: hidden = silu(X Gw^T) * (X Uw^T) ----------
__global__ __launch_bounds__(256) void gemm_gateup_kernel(
    const bf16_t* __restrict__ Xb, const bf16_t* __restrict__ Gw,
    const bf16_t* __restrict__ Uw, bf16_t* __restrict__ Hd) {
  __shared__ bf16_t As[128 * LDSS];
  __shared__ bf16_t Bgs[64 * LDSS];
  __shared__ bf16_t Bus[64 * LDSS];
  const int tid = threadIdx.x, lane = tid & 31, wave = tid >> 5;
  const int wm = (wave & 3) * 32, wn = (wave >> 2) * 32;
  const int lrow = lane & 15, lhalf = lane >> 4, k0 = lhalf * 8;
  const int tRow0 = blockIdx.y * 128, fCol0 = blockIdx.x * 64;

  v8f accG[2][2] = {};
  v8f accU[2][2] = {};

  for (int kb = 0; kb < H_DIM; kb += 32) {
#pragma unroll
    for (int it = 0; it < 2; ++it) {
      int chunk = tid + it * 256, r = chunk >> 2, c = chunk & 3;
      const bf16_t* src = Xb + (size_t)(tRow0 + r) * H_DIM + kb + c * 8;
      if (kb + 32 < H_DIM) __builtin_prefetch(src + 32, 0, 1);
      *(v8bf*)&As[r * LDSS + c * 8] = *(const v8bf*)src;
    }
    {
      int r = tid >> 2, c = tid & 3;
      const bf16_t* sg = Gw + (size_t)(fCol0 + r) * H_DIM + kb + c * 8;
      const bf16_t* su = Uw + (size_t)(fCol0 + r) * H_DIM + kb + c * 8;
      if (kb + 32 < H_DIM) { __builtin_prefetch(sg + 32, 0, 1); __builtin_prefetch(su + 32, 0, 1); }
      *(v8bf*)&Bgs[r * LDSS + c * 8] = *(const v8bf*)sg;
      *(v8bf*)&Bus[r * LDSS + c * 8] = *(const v8bf*)su;
    }
    __syncthreads();
    FragBF a[2], bg[2], bu[2];
#pragma unroll
    for (int mi = 0; mi < 2; ++mi) {
      int r = wm + mi * 16 + lrow;
      a[mi].h[0] = *(const v8bf*)&As[r * LDSS + k0];
      a[mi].h[1] = *(const v8bf*)&As[r * LDSS + k0 + 16];
    }
#pragma unroll
    for (int ni = 0; ni < 2; ++ni) {
      int r = wn + ni * 16 + lrow;
      bg[ni].h[0] = *(const v8bf*)&Bgs[r * LDSS + k0];
      bg[ni].h[1] = *(const v8bf*)&Bgs[r * LDSS + k0 + 16];
      bu[ni].h[0] = *(const v8bf*)&Bus[r * LDSS + k0];
      bu[ni].h[1] = *(const v8bf*)&Bus[r * LDSS + k0 + 16];
    }
#pragma unroll
    for (int mi = 0; mi < 2; ++mi)
#pragma unroll
      for (int ni = 0; ni < 2; ++ni) {
        accG[mi][ni] = __builtin_amdgcn_wmma_f32_16x16x32_bf16(
            false, a[mi].v, false, bg[ni].v, (short)0, accG[mi][ni], false, false);
        accU[mi][ni] = __builtin_amdgcn_wmma_f32_16x16x32_bf16(
            false, a[mi].v, false, bu[ni].v, (short)0, accU[mi][ni], false, false);
      }
    __syncthreads();
  }
#pragma unroll
  for (int mi = 0; mi < 2; ++mi)
#pragma unroll
    for (int ni = 0; ni < 2; ++ni) {
      int M = tRow0 + wm + mi * 16 + lhalf * 8;
      int N = fCol0 + wn + ni * 16 + lrow;
#pragma unroll
      for (int r = 0; r < 8; ++r) {
        float g = accG[mi][ni][r], u = accU[mi][ni][r];
        float h = (g / (1.f + __expf(-g))) * u;   // silu(g)*u
        Hd[(size_t)(M + r) * FF_DIM + N] = f2bf(h);
      }
    }
}

// ---------- WMMA GEMM 2: out = Hidden Dw^T (f32 out) ----------
__global__ __launch_bounds__(256) void gemm_down_kernel(
    const bf16_t* __restrict__ Hd, const bf16_t* __restrict__ Dw,
    float* __restrict__ Out) {
  __shared__ bf16_t As[128 * LDSS];
  __shared__ bf16_t Bs[64 * LDSS];
  const int tid = threadIdx.x, lane = tid & 31, wave = tid >> 5;
  const int wm = (wave & 3) * 32, wn = (wave >> 2) * 32;
  const int lrow = lane & 15, lhalf = lane >> 4, k0 = lhalf * 8;
  const int tRow0 = blockIdx.y * 128, hCol0 = blockIdx.x * 64;

  v8f acc[2][2] = {};

  for (int kb = 0; kb < FF_DIM; kb += 32) {
#pragma unroll
    for (int it = 0; it < 2; ++it) {
      int chunk = tid + it * 256, r = chunk >> 2, c = chunk & 3;
      const bf16_t* src = Hd + (size_t)(tRow0 + r) * FF_DIM + kb + c * 8;
      if (kb + 32 < FF_DIM) __builtin_prefetch(src + 32, 0, 1);
      *(v8bf*)&As[r * LDSS + c * 8] = *(const v8bf*)src;
    }
    {
      int r = tid >> 2, c = tid & 3;
      const bf16_t* sb = Dw + (size_t)(hCol0 + r) * FF_DIM + kb + c * 8;
      if (kb + 32 < FF_DIM) __builtin_prefetch(sb + 32, 0, 1);
      *(v8bf*)&Bs[r * LDSS + c * 8] = *(const v8bf*)sb;
    }
    __syncthreads();
    FragBF a[2], b[2];
#pragma unroll
    for (int mi = 0; mi < 2; ++mi) {
      int r = wm + mi * 16 + lrow;
      a[mi].h[0] = *(const v8bf*)&As[r * LDSS + k0];
      a[mi].h[1] = *(const v8bf*)&As[r * LDSS + k0 + 16];
    }
#pragma unroll
    for (int ni = 0; ni < 2; ++ni) {
      int r = wn + ni * 16 + lrow;
      b[ni].h[0] = *(const v8bf*)&Bs[r * LDSS + k0];
      b[ni].h[1] = *(const v8bf*)&Bs[r * LDSS + k0 + 16];
    }
#pragma unroll
    for (int mi = 0; mi < 2; ++mi)
#pragma unroll
      for (int ni = 0; ni < 2; ++ni)
        acc[mi][ni] = __builtin_amdgcn_wmma_f32_16x16x32_bf16(
            false, a[mi].v, false, b[ni].v, (short)0, acc[mi][ni], false, false);
    __syncthreads();
  }
#pragma unroll
  for (int mi = 0; mi < 2; ++mi)
#pragma unroll
    for (int ni = 0; ni < 2; ++ni) {
      int M = tRow0 + wm + mi * 16 + lhalf * 8;
      int N = hCol0 + wn + ni * 16 + lrow;
#pragma unroll
      for (int r = 0; r < 8; ++r)
        Out[(size_t)(M + r) * H_DIM + N] = acc[mi][ni][r];
    }
}

// ---------- launch ----------
static inline size_t align256(size_t x) { return (x + 255) & ~(size_t)255; }

extern "C" void kernel_launch(void* const* d_in, const int* in_sizes, int n_in,
                              void* d_out, int out_size, void* d_ws, size_t ws_size,
                              hipStream_t stream) {
  (void)in_sizes; (void)n_in; (void)out_size; (void)ws_size;
  const float* x     = (const float*)d_in[0];
  const float* gw    = (const float*)d_in[1];
  const float* uw    = (const float*)d_in[2];
  const float* dw    = (const float*)d_in[3];
  const float* rw    = (const float*)d_in[4];
  const float* ml    = (const float*)d_in[5];
  const float* atoms = (const float*)d_in[6];
  const float* eaw   = (const float*)d_in[7];
  const float* imp   = (const float*)d_in[8];
  float* out = (float*)d_out;
  char* ws = (char*)d_ws;

  size_t off = 0;
  size_t XB_OFF  = off; off += align256((size_t)T_TOK * H_DIM * 2);
  size_t WB_OFF  = off; off += align256((size_t)P_NUM * 2);
  size_t HD_OFF  = off; off += align256((size_t)T_TOK * FF_DIM * 2);
  size_t WD_OFF  = off; off += align256((size_t)E_NUM * NS_NUM * 4);
  size_t AW_OFF  = off; off += align256((size_t)E_NUM * A_NUM * 4);
  size_t MD_OFF  = off; off += align256((size_t)NS_NUM * 4);
  size_t SEL_OFF = off; off += (size_t)SEL_PAD * 8;
  size_t TIE_OFF = off; off += (size_t)TIE_CAP * 8;
  size_t H1_OFF  = off; off += 2048 * 4;
  size_t H2_OFF  = off; off += 2048 * 4;
  size_t ST_OFF  = off; off += 128;
  size_t CT_OFF  = off; off += 128;

  bf16_t* Xb = (bf16_t*)(ws + XB_OFF);
  bf16_t* Wb = (bf16_t*)(ws + WB_OFF);       // gate | up | down contiguous
  bf16_t* Hd = (bf16_t*)(ws + HD_OFF);
  float*  wd = (float*)(ws + WD_OFF);
  float*  aw = (float*)(ws + AW_OFF);
  float*  md = (float*)(ws + MD_OFF);
  unsigned long long* sel = (unsigned long long*)(ws + SEL_OFF);
  unsigned long long* tie = (unsigned long long*)(ws + TIE_OFF);
  unsigned* h1 = (unsigned*)(ws + H1_OFF);
  unsigned* h2 = (unsigned*)(ws + H2_OFF);
  float* stat = (float*)(ws + ST_OFF);
  int*   ctl  = (int*)(ws + CT_OFF);

  // zero histograms / stats / counters (contiguous region)
  hipMemsetAsync(ws + H1_OFF, 0, 2048 * 4 + 2048 * 4 + 128 + 128, stream);

  router_kernel<<<T_TOK / 8, 256, 0, stream>>>(x, rw, out, stat);
  aux_kernel<<<1, 1, 0, stream>>>(stat, out);

  hist1_kernel<<<2048, 256, 0, stream>>>(ml, h1);
  thresh1_kernel<<<1, 1, 0, stream>>>(h1, ctl);
  hist2_kernel<<<2048, 256, 0, stream>>>(ml, ctl, h2);
  thresh2_kernel<<<1, 1, 0, stream>>>(h2, ctl);
  select_kernel<<<2048, 256, 0, stream>>>(ml, ctl, sel, tie);
  tie_sort_kernel<<<1, 1024, 0, stream>>>(tie, sel, ctl);
  sel_sort_kernel<<<1, 1024, 0, stream>>>(sel);

  atom_softmax_kernel<<<1, 64, 0, stream>>>(eaw, aw);
  wd_kernel<<<(E_NUM * NS_NUM + 255) / 256, 256, 0, stream>>>(atoms, aw, imp, wd);
  md_kernel<<<(NS_NUM + 255) / 256, 256, 0, stream>>>(wd, stat, md);

  convert_x_kernel<<<4096, 256, 0, stream>>>(x, Xb);
  convert_w_kernel<<<8192, 256, 0, stream>>>(gw, uw, dw, Wb);
  fixup_kernel<<<(NS_NUM + 255) / 256, 256, 0, stream>>>(sel, md, gw, uw, dw, Wb);

  dim3 g1(FF_DIM / 64, T_TOK / 128);
  gemm_gateup_kernel<<<g1, 256, 0, stream>>>(Xb, Wb, Wb + (size_t)N1, Hd);
  dim3 g2(H_DIM / 64, T_TOK / 128);
  gemm_down_kernel<<<g2, 256, 0, stream>>>(Hd, Wb + (size_t)2 * N1, out);
}
